// SelfAttention_79087527788783
// MI455X (gfx1250) — compile-verified, matrix operand order checked by reading, not run on hardware
//
#include <hip/hip_runtime.h>

typedef __attribute__((ext_vector_type(16))) _Float16 v16h;
typedef __attribute__((ext_vector_type(8)))  _Float16 v8h;
typedef __attribute__((ext_vector_type(8)))  float    v8f;

#define WMMA_F16(A, B, C) \
  __builtin_amdgcn_wmma_f32_16x16x32_f16(false, (A), false, (B), (short)0, (C), false, false)

static constexpr int   Bn   = 8;
static constexpr int   Ln   = 2048;
static constexpr int   Dn   = 64;
static constexpr float NEGV = -1e9f;

// ---------------------------------------------------------------------------
// Pre-pass: K -> f16 row-major, V -> f16 transposed [B][D][L].
// (Q is converted in-kernel: it is read only once per wave.)
// ---------------------------------------------------------------------------
__global__ __launch_bounds__(256) void preconvert(const float* __restrict__ K,
                                                  const float* __restrict__ V,
                                                  _Float16* __restrict__ Kh,
                                                  _Float16* __restrict__ Vt) {
  const int idx = blockIdx.x * 256 + threadIdx.x;   // 2^20 elements
  Kh[idx] = (_Float16)K[idx];
  const int d = idx & (Dn - 1);
  const int l = (idx >> 6) & (Ln - 1);
  const int b = idx >> 17;                          // Ln*Dn = 2^17
  Vt[((size_t)b * Dn + d) * Ln + l] = (_Float16)V[idx];
}

// 16-element f16 fragment stripe: elements 0..7 at p+hl*8, 8..15 at p+16+hl*8.
__device__ __forceinline__ v16h frag_f16(const _Float16* __restrict__ p, int hl) {
  const v8h lo = *(const v8h*)(p + hl * 8);
  const v8h hi = *(const v8h*)(p + 16 + hl * 8);
  v16h a;
#pragma unroll
  for (int i = 0; i < 8; ++i) { a[i] = lo[i]; a[8 + i] = hi[i]; }
  return a;
}

// f32 -> f16 fragment (used only for the one-time Q conversion / fallback).
__device__ __forceinline__ v16h frag_rowchunk(const float* __restrict__ p, int hl, float scale) {
  v16h a;
  const float* p0 = p + hl * 8;
  const float* p1 = p + 16 + hl * 8;
#pragma unroll
  for (int i = 0; i < 8; ++i) a[i] = (_Float16)(p0[i] * scale);
#pragma unroll
  for (int i = 0; i < 8; ++i) a[8 + i] = (_Float16)(p1[i] * scale);
  return a;
}

// S^T for one 64-key block: 4 C-layout subtiles.
// T[s], VGPR r, lane (ln,hl) == S[q=qb+ln][key = kb + 16*s + r + 8*hl].
__device__ __forceinline__ void score_block(const _Float16* __restrict__ Khb, int kb,
                                            v16h qbf0, v16h qbf1,
                                            int ln, int hl, v8f T[4]) {
  const v8f zero = {};
#pragma unroll
  for (int s = 0; s < 4; ++s) {
    const _Float16* kr = Khb + (size_t)(kb + 16 * s + ln) * Dn;
    v8f t = zero;
    t = WMMA_F16(frag_f16(kr, hl), qbf0, t);
    t = WMMA_F16(frag_f16(kr + 32, hl), qbf1, t);
    T[s] = t;
  }
}

// Online-softmax + P*V. MODE: 0 = no mask, 1 = penalize j<i (fw diagonal),
// 2 = penalize j>i (bw diagonal). The exp'd S^T tile is directly the two
// A-fragments of the two 32-key P*V WMMAs. V fragments are loaded here so
// their registers are not live across the score/softmax phase.
template <int MODE>
__device__ __forceinline__ void flash_update64(const v8f T[4], int kb, int i_q,
                                               float& m, float& l, v8f acc[4],
                                               const _Float16* __restrict__ vbase,
                                               int ln, int hl) {
  float p[4][8];
  float mx = -3.0e38f;
#pragma unroll
  for (int s = 0; s < 4; ++s)
#pragma unroll
    for (int r = 0; r < 8; ++r) {
      float sv = T[s][r];
      if (MODE == 1) { if (kb + 16 * s + r + 8 * hl < i_q) sv += NEGV; }
      if (MODE == 2) { if (kb + 16 * s + r + 8 * hl > i_q) sv += NEGV; }
      p[s][r] = sv;
      mx = fmaxf(mx, sv);
    }
  mx = fmaxf(mx, __shfl_xor(mx, 16));

  const float mn = fmaxf(m, mx);
  const float sf = __expf(m - mn);
  m = mn;

  float rs = 0.f;
#pragma unroll
  for (int s = 0; s < 4; ++s)
#pragma unroll
    for (int r = 0; r < 8; ++r) { p[s][r] = __expf(p[s][r] - mn); rs += p[s][r]; }
  rs += __shfl_xor(rs, 16);
  l = l * sf + rs;

  v16h pa0, pa1;
#pragma unroll
  for (int i = 0; i < 8; ++i) {
    pa0[i] = (_Float16)p[0][i]; pa0[8 + i] = (_Float16)p[1][i];
    pa1[i] = (_Float16)p[2][i]; pa1[8 + i] = (_Float16)p[3][i];
  }

#pragma unroll
  for (int r = 0; r < 8; ++r) {
    const float sfr = __shfl(sf, r + 8 * hl);
#pragma unroll
    for (int t = 0; t < 4; ++t) acc[t][r] *= sfr;
  }

#pragma unroll
  for (int t = 0; t < 4; ++t) {
    const _Float16* vrow = vbase + (size_t)(t * 16 + ln) * Ln;
    acc[t] = WMMA_F16(pa0, frag_f16(vrow, hl), acc[t]);
    acc[t] = WMMA_F16(pa1, frag_f16(vrow + 32, hl), acc[t]);
  }
}

__device__ __forceinline__ void prefetch_block(const _Float16* __restrict__ Khb,
                                               const _Float16* __restrict__ Vtb,
                                               int kb, int lane) {
  __builtin_prefetch(Khb + (size_t)(kb + lane) * Dn, 0, 3);        // 128B/row
  __builtin_prefetch(Khb + (size_t)(kb + 32 + lane) * Dn, 0, 3);
  __builtin_prefetch(Vtb + (size_t)lane * Ln + kb, 0, 3);
  __builtin_prefetch(Vtb + (size_t)(lane + 32) * Ln + kb, 0, 3);
}

__global__ __launch_bounds__(128) void bidir_flash_attn_f16(
    const float* __restrict__ Q, const _Float16* __restrict__ Kh,
    const _Float16* __restrict__ Vt, float* __restrict__ O) {
  const int lane = threadIdx.x & 31;
  const int wid  = threadIdx.x >> 5;
  const int ln   = lane & 15;
  const int hl   = lane >> 4;

  const int tile = blockIdx.x * 4 + wid;   // 1024 q-tiles total
  const int b    = tile >> 7;              // L/16 = 128 tiles per batch
  const int qb   = (tile & 127) << 4;

  const _Float16* Khb = Kh + (size_t)b * Ln * Dn;
  const _Float16* Vtb = Vt + (size_t)b * Dn * Ln;

  // One-time Q^T B-fragments from f32, pre-scaled by 1/sqrt(D).
  const float* qrow = Q + ((size_t)b * Ln + qb + ln) * Dn;
  const v16h qbf0 = frag_rowchunk(qrow, hl, 0.125f);
  const v16h qbf1 = frag_rowchunk(qrow + 32, hl, 0.125f);

  const v8f zero = {};
  float m_f = -1e30f, l_f = 0.f, m_b = -1e30f, l_b = 0.f;
  v8f acc_f[4], acc_b[4];
#pragma unroll
  for (int t = 0; t < 4; ++t) { acc_f[t] = zero; acc_b[t] = zero; }

  const int i_q   = qb + ln;
  const int kdiag = qb & ~63;              // the single diagonal 64-key block

  // Phase 1: keys strictly below the diagonal block -> bw only, no mask.
  for (int kb = 0; kb < kdiag; kb += 64) {
    prefetch_block(Khb, Vtb, kb + 64, lane);
    v8f T[4];
    score_block(Khb, kb, qbf0, qbf1, ln, hl, T);
    flash_update64<0>(T, kb, i_q, m_b, l_b, acc_b, Vtb + kb, ln, hl);
  }

  // Phase 2: the diagonal block -> both directions with elementwise masks.
  {
    if (kdiag + 64 < Ln) prefetch_block(Khb, Vtb, kdiag + 64, lane);
    v8f T[4];
    score_block(Khb, kdiag, qbf0, qbf1, ln, hl, T);
    flash_update64<1>(T, kdiag, i_q, m_f, l_f, acc_f, Vtb + kdiag, ln, hl);
    flash_update64<2>(T, kdiag, i_q, m_b, l_b, acc_b, Vtb + kdiag, ln, hl);
  }

  // Phase 3: keys strictly above the diagonal block -> fw only, no mask.
  for (int kb = kdiag + 64; kb < Ln; kb += 64) {
    if (kb + 64 < Ln) prefetch_block(Khb, Vtb, kb + 64, lane);
    v8f T[4];
    score_block(Khb, kb, qbf0, qbf1, ln, hl, T);
    flash_update64<0>(T, kb, i_q, m_f, l_f, acc_f, Vtb + kb, ln, hl);
  }

  float* Ob = O + ((size_t)b * Ln + qb) * Dn;
  const float inv_f = 1.0f / l_f;
  const float inv_b = 1.0f / l_b;
#pragma unroll
  for (int r = 0; r < 8; ++r) {
    const float rf = __shfl(inv_f, r + 8 * hl);
    const float rb = __shfl(inv_b, r + 8 * hl);
#pragma unroll
    for (int t = 0; t < 4; ++t)
      Ob[(size_t)(r + 8 * hl) * Dn + t * 16 + ln] = acc_f[t][r] * rf + acc_b[t][r] * rb;
  }
}

// ---------------------------------------------------------------------------
// Fallback (workspace too small): self-contained f32-input version (BK=32).
// ---------------------------------------------------------------------------
__device__ __forceinline__ v16h frag_colstride(const float* __restrict__ p, int hl) {
  v16h a;
#pragma unroll
  for (int i = 0; i < 8; ++i) a[i] = (_Float16)p[(hl * 8 + i) * Dn];
#pragma unroll
  for (int i = 0; i < 8; ++i) a[8 + i] = (_Float16)p[(16 + hl * 8 + i) * Dn];
  return a;
}

__device__ __forceinline__ void flash_update32(v8f T0, v8f T1,
                                               float& m, float& l, v8f acc[4],
                                               const v16h* __restrict__ vf, int hl) {
  float mx = fmaxf(T0[0], T1[0]);
#pragma unroll
  for (int r = 1; r < 8; ++r) mx = fmaxf(mx, fmaxf(T0[r], T1[r]));
  mx = fmaxf(mx, __shfl_xor(mx, 16));
  const float mn = fmaxf(m, mx);
  const float sf = __expf(m - mn);
  m = mn;
  float p0[8], p1[8];
  float rs = 0.f;
#pragma unroll
  for (int r = 0; r < 8; ++r) {
    p0[r] = __expf(T0[r] - mn);
    p1[r] = __expf(T1[r] - mn);
    rs += p0[r] + p1[r];
  }
  rs += __shfl_xor(rs, 16);
  l = l * sf + rs;
  v16h pa;
#pragma unroll
  for (int i = 0; i < 8; ++i) { pa[i] = (_Float16)p0[i]; pa[8 + i] = (_Float16)p1[i]; }
#pragma unroll
  for (int r = 0; r < 8; ++r) {
    const float sfr = __shfl(sf, r + 8 * hl);
#pragma unroll
    for (int t = 0; t < 4; ++t) acc[t][r] *= sfr;
  }
#pragma unroll
  for (int t = 0; t < 4; ++t) acc[t] = WMMA_F16(pa, vf[t], acc[t]);
}

__global__ __launch_bounds__(128) void bidir_flash_attn_f32(
    const float* __restrict__ Q, const float* __restrict__ K,
    const float* __restrict__ V, float* __restrict__ O) {
  const int lane = threadIdx.x & 31;
  const int wid  = threadIdx.x >> 5;
  const int ln   = lane & 15;
  const int hl   = lane >> 4;
  const int tile = blockIdx.x * 4 + wid;
  const int b    = tile >> 7;
  const int qb   = (tile & 127) << 4;

  const float* Qb = Q + ((size_t)b * Ln + qb) * Dn;
  const float* Kb = K + (size_t)b * Ln * Dn;
  const float* Vb = V + (size_t)b * Ln * Dn;

  const float* qrow = Qb + ln * Dn;
  const v16h qbf0 = frag_rowchunk(qrow, hl, 0.125f);
  const v16h qbf1 = frag_rowchunk(qrow + 32, hl, 0.125f);

  const v8f zero = {};
  float m_f = -1e30f, l_f = 0.f, m_b = -1e30f, l_b = 0.f;
  v8f acc_f[4], acc_b[4];
#pragma unroll
  for (int t = 0; t < 4; ++t) { acc_f[t] = zero; acc_b[t] = zero; }
  const int i_q = qb + ln;

  for (int kb = 0; kb < Ln; kb += 32) {
    const bool below = (kb + 31) < qb;
    const bool above = kb > (qb + 15);
    const bool diag  = !below && !above;

    const float* kr0 = Kb + (size_t)(kb + ln) * Dn;
    const float* kr1 = Kb + (size_t)(kb + 16 + ln) * Dn;
    v16h ka00 = frag_rowchunk(kr0, hl, 1.0f);
    v16h ka01 = frag_rowchunk(kr0 + 32, hl, 1.0f);
    v16h ka10 = frag_rowchunk(kr1, hl, 1.0f);
    v16h ka11 = frag_rowchunk(kr1 + 32, hl, 1.0f);

    v8f T0 = zero, T1 = zero;
    T0 = WMMA_F16(ka00, qbf0, T0);
    T0 = WMMA_F16(ka01, qbf1, T0);
    T1 = WMMA_F16(ka10, qbf0, T1);
    T1 = WMMA_F16(ka11, qbf1, T1);

    v16h vf[4];
#pragma unroll
    for (int t = 0; t < 4; ++t)
      vf[t] = frag_colstride(Vb + (size_t)kb * Dn + t * 16 + ln, hl);

    if (!below) {
      v8f F0 = T0, F1 = T1;
      if (diag) {
#pragma unroll
        for (int r = 0; r < 8; ++r) {
          const int j0 = kb + r + 8 * hl;
          if (j0 < i_q)      F0[r] += NEGV;
          if (j0 + 16 < i_q) F1[r] += NEGV;
        }
      }
      flash_update32(F0, F1, m_f, l_f, acc_f, vf, hl);
    }
    if (!above) {
      v8f B0 = T0, B1 = T1;
      if (diag) {
#pragma unroll
        for (int r = 0; r < 8; ++r) {
          const int j0 = kb + r + 8 * hl;
          if (j0 > i_q)      B0[r] += NEGV;
          if (j0 + 16 > i_q) B1[r] += NEGV;
        }
      }
      flash_update32(B0, B1, m_b, l_b, acc_b, vf, hl);
    }
  }

  float* Ob = O + ((size_t)b * Ln + qb) * Dn;
  const float inv_f = 1.0f / l_f;
  const float inv_b = 1.0f / l_b;
#pragma unroll
  for (int r = 0; r < 8; ++r) {
    const float rf = __shfl(inv_f, r + 8 * hl);
    const float rb = __shfl(inv_b, r + 8 * hl);
#pragma unroll
    for (int t = 0; t < 4; ++t)
      Ob[(size_t)(r + 8 * hl) * Dn + t * 16 + ln] = acc_f[t][r] * rf + acc_b[t][r] * rb;
  }
}

extern "C" void kernel_launch(void* const* d_in, const int* in_sizes, int n_in,
                              void* d_out, int out_size, void* d_ws, size_t ws_size,
                              hipStream_t stream) {
  (void)in_sizes; (void)n_in; (void)out_size;
  const float* q = (const float*)d_in[0];
  const float* k = (const float*)d_in[1];
  const float* v = (const float*)d_in[2];
  float* o = (float*)d_out;

  const size_t elems = (size_t)Bn * Ln * Dn;            // 2^20
  const size_t need  = elems * sizeof(_Float16) * 2;    // 4 MB (Kh + Vt)

  if (d_ws != nullptr && ws_size >= need) {
    _Float16* Kh = (_Float16*)d_ws;
    _Float16* Vt = Kh + elems;
    preconvert<<<(int)(elems / 256), 256, 0, stream>>>(k, v, Kh, Vt);
    bidir_flash_attn_f16<<<256, 128, 0, stream>>>(q, Kh, Vt, o);
  } else {
    bidir_flash_attn_f32<<<256, 128, 0, stream>>>(q, k, v, o);
  }
}
